// SiRNAGenerator_40510131536342
// MI455X (gfx1250) — matrix operand order repaired and checked by reading, not compiled, not run on hardware
//
#include <hip/hip_runtime.h>
#include <hip/hip_bf16.h>
#include <math.h>

#define Nn   20000
#define Ee   160000
#define Bb   1000
#define Hh   2
#define Cc   128
#define Dd   256
#define EMBd 64
#define Vv   5
#define Tt   21
#define SOS  1
#define INV_SQRT_C 0.08838834764831845f  // 1/sqrt(128)

typedef __attribute__((ext_vector_type(16))) _Float16 v16h;
typedef __attribute__((ext_vector_type(8)))  _Float16 v8h;
typedef __attribute__((ext_vector_type(4)))  _Float16 v4h;
typedef __attribute__((ext_vector_type(8)))  float    v8f;

// ---------- helpers ----------
__device__ __forceinline__ int f2o(float f) {           // order-preserving float->int
    int i = __float_as_int(f);
    return i >= 0 ? i : (i ^ 0x7fffffff);
}
__device__ __forceinline__ float o2f(int i) {
    return __int_as_float(i >= 0 ? i : (i ^ 0x7fffffff));
}
__device__ __forceinline__ float wave_sum(float p) {
    #pragma unroll
    for (int off = 16; off > 0; off >>= 1) p += __shfl_xor(p, off, 32);
    return p;
}
__device__ __forceinline__ float sigmoidf(float x) { return 1.0f / (1.0f + expf(-x)); }

// ---------- precision prep: f32 -> f16 activations (vectorized) ----------
__global__ void conv_f16(const float* __restrict__ src, _Float16* __restrict__ dst, int n4) {
    int i = blockIdx.x * blockDim.x + threadIdx.x;
    if (i >= n4) return;
    float4 v = ((const float4*)src)[i];
    v4h h = { (_Float16)v.x, (_Float16)v.y, (_Float16)v.z, (_Float16)v.w };
    ((v4h*)dst)[i] = h;
}

// ---------- precision prep: W[256,256] f32 row-major -> f16 col-major ----------
__global__ void transpose_w_f16(const float* __restrict__ W, _Float16* __restrict__ Wt) {
    int i = blockIdx.x * blockDim.x + threadIdx.x;   // 65536 threads
    int k = i >> 8, n = i & 255;
    Wt[(size_t)n * 256 + k] = (_Float16)W[(size_t)k * 256 + n];
}

// ---------- WMMA GEMM: Y[M,256] = Xh[M,256](f16) @ Wt(f16,col-major) + bias ----------
// One wave computes a 16x16 output tile; 8 waves/block; grid = (M/16, 2).
// K loop fully unrolled -> 8 v_wmma per wave, operands via b128 loads.
__global__ void gemm256_wmma(const _Float16* __restrict__ Xh, const _Float16* __restrict__ Wt,
                             const float* __restrict__ bias, float* __restrict__ Y) {
    const int lane    = threadIdx.x & 31;
    const int wave    = threadIdx.x >> 5;
    const int rowTile = blockIdx.x;
    const int colTile = blockIdx.y * 8 + wave;       // 0..15
    const int row     = rowTile * 16 + (lane & 15);
    const int col     = colTile * 16 + (lane & 15);
    const int kbA     = (lane < 16) ? 0 : 16;        // A: contiguous K per lane-half
    const int kbB     = (lane < 16) ? 0 : 16;        // B: contiguous K per lane-half
    const _Float16* xr = Xh + (size_t)row * 256 + kbA;
    const _Float16* wc = Wt + (size_t)col * 256 + kbB;

    v8f acc = {};
    #pragma unroll
    for (int k0 = 0; k0 < 256; k0 += 32) {
        v8h alo = *(const v8h*)(xr + k0);            // K = kbA + 0..7
        v8h ahi = *(const v8h*)(xr + k0 + 8);        // K = kbA + 8..15
        v16h a;
        #pragma unroll
        for (int j = 0; j < 8; ++j) { a[j] = alo[j]; a[8 + j] = ahi[j]; }
        v16h b = *(const v16h*)(wc + k0);            // 16 contiguous K values
        acc = __builtin_amdgcn_wmma_f32_16x16x32_f16(false, a, false, b,
                                                     (short)0, acc, false, false);
    }
    const float bc    = bias[col];
    const int   rbase = rowTile * 16 + ((lane < 16) ? 0 : 8);
    #pragma unroll
    for (int r = 0; r < 8; ++r)
        Y[(size_t)(rbase + r) * 256 + col] = acc[r] + bc;   // C/D layout: VGPR r -> M=r / M=8+r
}

// ---------- layer 0: X[N,4] -> Q,K,V and S(into accum), all [N,256] ----------
__global__ void layer0_qkvs(const float* __restrict__ x,
                            const float* __restrict__ wq, const float* __restrict__ bq,
                            const float* __restrict__ wk, const float* __restrict__ bk,
                            const float* __restrict__ wv, const float* __restrict__ bv,
                            const float* __restrict__ wsk, const float* __restrict__ bsk,
                            float* __restrict__ Q, float* __restrict__ K,
                            float* __restrict__ V, float* __restrict__ acc) {
    int i = blockIdx.x * blockDim.x + threadIdx.x;
    if (i >= Nn * 256) return;
    int n = i >> 8, c = i & 255;
    float x0 = x[n * 4 + 0], x1 = x[n * 4 + 1], x2 = x[n * 4 + 2], x3 = x[n * 4 + 3];
    Q[i]   = x0 * wq[c]  + x1 * wq[256 + c]  + x2 * wq[512 + c]  + x3 * wq[768 + c]  + bq[c];
    K[i]   = x0 * wk[c]  + x1 * wk[256 + c]  + x2 * wk[512 + c]  + x3 * wk[768 + c]  + bk[c];
    V[i]   = x0 * wv[c]  + x1 * wv[256 + c]  + x2 * wv[512 + c]  + x3 * wv[768 + c]  + bv[c];
    acc[i] = x0 * wsk[c] + x1 * wsk[256 + c] + x2 * wsk[512 + c] + x3 * wsk[768 + c] + bsk[c];
}

// ---------- attention: init, scores+max, exp+sum, scatter ----------
__global__ void attn_init(int* __restrict__ segmaxi, float* __restrict__ segsum) {
    int i = blockIdx.x * blockDim.x + threadIdx.x;
    if (i >= Nn * Hh) return;
    segmaxi[i] = f2o(-3.0e38f);
    segsum[i]  = 0.0f;
}

__global__ void edge_scores(const float* __restrict__ Q, const float* __restrict__ K,
                            const int* __restrict__ src, const int* __restrict__ tgt,
                            float* __restrict__ score, int* __restrict__ segmaxi) {
    int lane = threadIdx.x & 31;
    int e = (blockIdx.x * blockDim.x + threadIdx.x) >> 5;
    if (e >= Ee) return;
    int s = src[e], t = tgt[e];
    #pragma unroll
    for (int h = 0; h < Hh; ++h) {
        const float4 qv = *(const float4*)(Q + (size_t)t * 256 + h * 128 + lane * 4);
        const float4 kv = *(const float4*)(K + (size_t)s * 256 + h * 128 + lane * 4);
        float p = qv.x * kv.x + qv.y * kv.y + qv.z * kv.z + qv.w * kv.w;
        p = wave_sum(p);
        if (lane == 0) {
            float sc = p * INV_SQRT_C;
            score[e * 2 + h] = sc;
            atomicMax(&segmaxi[t * 2 + h], f2o(sc));
        }
    }
}

__global__ void edge_exp_sum(float* __restrict__ score, const int* __restrict__ tgt,
                             const int* __restrict__ segmaxi, float* __restrict__ segsum) {
    int i = blockIdx.x * blockDim.x + threadIdx.x;
    if (i >= Ee * Hh) return;
    int e = i >> 1, h = i & 1;
    int t = tgt[e];
    float a = expf(score[i] - o2f(segmaxi[t * 2 + h]));
    score[i] = a;
    atomicAdd(&segsum[t * 2 + h], a);
}

__global__ void edge_scatter(const float* __restrict__ score, const float* __restrict__ segsum,
                             const float* __restrict__ V, const int* __restrict__ src,
                             const int* __restrict__ tgt, float* __restrict__ acc) {
    int i = blockIdx.x * blockDim.x + threadIdx.x;   // (edge, head, 4-ch chunk)
    if (i >= Ee * Hh * 32) return;
    int e = i >> 6, rem = i & 63, h = rem >> 5, c4 = (rem & 31) * 4;
    int s = src[e], t = tgt[e];
    float alpha = score[e * 2 + h] / segsum[t * 2 + h];
    const float4 vv = *(const float4*)(V + (size_t)s * 256 + h * 128 + c4);
    float* o = acc + (size_t)t * 256 + h * 128 + c4;
    atomicAdd(o + 0, alpha * vv.x);
    atomicAdd(o + 1, alpha * vv.y);
    atomicAdd(o + 2, alpha * vv.z);
    atomicAdd(o + 3, alpha * vv.w);
}

// ---------- fused ReLU + LayerNorm: hbuf = LN(relu(acc)) ; wave per row ----------
__global__ void relu_ln(const float* __restrict__ acc, const float* __restrict__ g,
                        const float* __restrict__ b, float* __restrict__ out) {
    int lane = threadIdx.x & 31;
    int row  = blockIdx.x * 8 + (threadIdx.x >> 5);
    const float* r = acc + (size_t)row * 256 + lane * 8;
    float v[8], s = 0.0f, s2 = 0.0f;
    #pragma unroll
    for (int j = 0; j < 8; ++j) {
        float x = r[j];
        x = x > 0.0f ? x : 0.0f;
        v[j] = x; s += x; s2 += x * x;
    }
    s  = wave_sum(s);
    s2 = wave_sum(s2);
    float mu   = s * (1.0f / 256.0f);
    float var  = s2 * (1.0f / 256.0f) - mu * mu;
    float rstd = rsqrtf(var + 1e-5f);
    float* o = out + (size_t)row * 256 + lane * 8;
    #pragma unroll
    for (int j = 0; j < 8; ++j) {
        int c = lane * 8 + j;
        o[j] = (v[j] - mu) * rstd * g[c] + b[c];
    }
}

// ---------- global mean pool + gfeat [B,258] (20 contiguous nodes per graph) ----------
__global__ void pool_gfeat(const float* __restrict__ h, const float* __restrict__ gc,
                           const float* __restrict__ sl, float* __restrict__ gfeat) {
    int g = blockIdx.x, c = threadIdx.x;
    float s = 0.0f;
    #pragma unroll 4
    for (int i = 0; i < 20; ++i) s += h[(size_t)(g * 20 + i) * 256 + c];
    gfeat[g * 258 + c] = s * (1.0f / 20.0f);
    if (c == 0) {
        gfeat[g * 258 + 256] = gc[g];
        gfeat[g * 258 + 257] = sl[g];
    }
}

// ---------- encoder FC: enc[B,64] = relu(gfeat @ enc_w + enc_b) ----------
__global__ void encoder_fc(const float* __restrict__ gfeat, const float* __restrict__ w,
                           const float* __restrict__ b, float* __restrict__ enc) {
    int g = blockIdx.x, j = threadIdx.x;
    float a = b[j];
    const float* gr = gfeat + g * 258;
    for (int k = 0; k < 258; ++k) a += gr[k] * w[k * 64 + j];
    enc[g * 64 + j] = a > 0.0f ? a : 0.0f;
}

// ---------- greedy LSTM decode: one block per graph, T steps in-block ----------
__global__ void lstm_decode(const float* __restrict__ enc, const float* __restrict__ emb,
                            const float* __restrict__ w_ih, const float* __restrict__ w_hh,
                            const float* __restrict__ b_ih, const float* __restrict__ b_hh,
                            const float* __restrict__ fc_w, const float* __restrict__ fc_b,
                            float* __restrict__ out) {
    __shared__ float s_inp[64], s_h[64], s_c[64], s_g[256], s_log[5];
    __shared__ int s_tok;
    int b = blockIdx.x, t = threadIdx.x;
    if (t < 64) {
        s_inp[t] = emb[SOS * 64 + t];
        s_h[t]   = enc[b * 64 + t];
        s_c[t]   = 0.0f;
    }
    __syncthreads();
    for (int step = 0; step < Tt; ++step) {
        float acc = b_ih[t] + b_hh[t];
        for (int k = 0; k < 64; ++k)
            acc += s_inp[k] * w_ih[k * 256 + t] + s_h[k] * w_hh[k * 256 + t];
        s_g[t] = acc;
        __syncthreads();
        if (t < 64) {
            float ig = sigmoidf(s_g[t]);
            float fg = sigmoidf(s_g[64 + t]);
            float gg = tanhf(s_g[128 + t]);
            float og = sigmoidf(s_g[192 + t]);
            float c  = fg * s_c[t] + ig * gg;
            s_c[t] = c;
            s_h[t] = og * tanhf(c);
        }
        __syncthreads();
        if (t < Vv) {
            float a = fc_b[t];
            for (int k = 0; k < 64; ++k) a += s_h[k] * fc_w[k * 5 + t];
            s_log[t] = a;
            out[(size_t)b * Tt * Vv + step * Vv + t] = a;
        }
        __syncthreads();
        if (t == 0) {
            int best = 0; float bv = s_log[0];
            #pragma unroll
            for (int v = 1; v < Vv; ++v) if (s_log[v] > bv) { bv = s_log[v]; best = v; }
            s_tok = best;
        }
        __syncthreads();
        if (t < 64) s_inp[t] = emb[s_tok * 64 + t];
        __syncthreads();
    }
}

// ================= host-side orchestration =================
extern "C" void kernel_launch(void* const* d_in, const int* in_sizes, int n_in,
                              void* d_out, int out_size, void* d_ws, size_t ws_size,
                              hipStream_t stream) {
    const float* x       = (const float*)d_in[0];
    const int*   eidx    = (const int*)d_in[1];
    const int*   src     = eidx;
    const int*   tgt     = eidx + Ee;
    const float* gc      = (const float*)d_in[3];
    const float* sl      = (const float*)d_in[4];
    const float* c0_wq   = (const float*)d_in[5];
    const float* c0_bq   = (const float*)d_in[6];
    const float* c0_wk   = (const float*)d_in[7];
    const float* c0_bk   = (const float*)d_in[8];
    const float* c0_wv   = (const float*)d_in[9];
    const float* c0_bv   = (const float*)d_in[10];
    const float* c0_ws   = (const float*)d_in[11];
    const float* c0_bs   = (const float*)d_in[12];
    const float* cw_q    = (const float*)d_in[13];
    const float* cb_q    = (const float*)d_in[14];
    const float* cw_k    = (const float*)d_in[15];
    const float* cb_k    = (const float*)d_in[16];
    const float* cw_v    = (const float*)d_in[17];
    const float* cb_v    = (const float*)d_in[18];
    const float* cw_s    = (const float*)d_in[19];
    const float* cb_s    = (const float*)d_in[20];
    const float* ln_g    = (const float*)d_in[21];
    const float* ln_b    = (const float*)d_in[22];
    const float* enc_w   = (const float*)d_in[23];
    const float* enc_b   = (const float*)d_in[24];
    const float* emb     = (const float*)d_in[25];
    const float* w_ih    = (const float*)d_in[26];
    const float* w_hh    = (const float*)d_in[27];
    const float* b_ih    = (const float*)d_in[28];
    const float* b_hh    = (const float*)d_in[29];
    const float* fc_w    = (const float*)d_in[30];
    const float* fc_b    = (const float*)d_in[31];
    float* out = (float*)d_out;

    // workspace layout
    const size_t NF = (size_t)Nn * 256;   // 5,120,000 floats
    float* ws      = (float*)d_ws;
    float* hbuf    = ws;
    float* qbuf    = ws + 1 * NF;
    float* kbuf    = ws + 2 * NF;
    float* vbuf    = ws + 3 * NF;
    float* accum   = ws + 4 * NF;         // S output + attention scatter target
    float* score   = ws + 5 * NF;         // [E,2]
    float* segsum  = score + (size_t)Ee * 2;
    int*   segmaxi = (int*)(segsum + (size_t)Nn * 2);
    float* gfeat   = (float*)(segmaxi + (size_t)Nn * 2);
    float* encb    = gfeat + (size_t)Bb * 258;
    _Float16* hx   = (_Float16*)(encb + (size_t)Bb * 64);     // [N,256] f16
    _Float16* wt   = hx + NF;                                  // [4][256][256] f16 col-major

    const dim3 gemmGrid(Nn / 16, 2);
    const int  blk = 256;

    auto run_attention = [&](const float* lg, const float* lb) {
        attn_init<<<(Nn * Hh + blk - 1) / blk, blk, 0, stream>>>(segmaxi, segsum);
        edge_scores<<<Ee / 8, blk, 0, stream>>>(qbuf, kbuf, src, tgt, score, segmaxi);
        edge_exp_sum<<<(Ee * Hh + blk - 1) / blk, blk, 0, stream>>>(score, tgt, segmaxi, segsum);
        edge_scatter<<<(Ee * Hh * 32 + blk - 1) / blk, blk, 0, stream>>>(score, segsum, vbuf,
                                                                         src, tgt, accum);
        relu_ln<<<Nn / 8, blk, 0, stream>>>(accum, lg, lb, hbuf);
    };

    // ---- layer 0 (in=4): scalar projections, then attention ----
    layer0_qkvs<<<(Nn * 256 + blk - 1) / blk, blk, 0, stream>>>(
        x, c0_wq, c0_bq, c0_wk, c0_bk, c0_wv, c0_bv, c0_ws, c0_bs,
        qbuf, kbuf, vbuf, accum);
    run_attention(ln_g + 0 * 256, ln_b + 0 * 256);

    // ---- layers 1..2 (in=256): f16 prep + WMMA GEMMs, then attention ----
    for (int i = 0; i < 2; ++i) {
        const size_t wo = (size_t)i * 256 * 256, bo = (size_t)i * 256;
        conv_f16<<<(int)(NF / 4 + blk - 1) / blk, blk, 0, stream>>>(hbuf, hx, (int)(NF / 4));
        transpose_w_f16<<<256, blk, 0, stream>>>(cw_q + wo, wt + 0 * 65536);
        transpose_w_f16<<<256, blk, 0, stream>>>(cw_k + wo, wt + 1 * 65536);
        transpose_w_f16<<<256, blk, 0, stream>>>(cw_v + wo, wt + 2 * 65536);
        transpose_w_f16<<<256, blk, 0, stream>>>(cw_s + wo, wt + 3 * 65536);
        gemm256_wmma<<<gemmGrid, blk, 0, stream>>>(hx, wt + 0 * 65536, cb_q + bo, qbuf);
        gemm256_wmma<<<gemmGrid, blk, 0, stream>>>(hx, wt + 1 * 65536, cb_k + bo, kbuf);
        gemm256_wmma<<<gemmGrid, blk, 0, stream>>>(hx, wt + 2 * 65536, cb_v + bo, vbuf);
        gemm256_wmma<<<gemmGrid, blk, 0, stream>>>(hx, wt + 3 * 65536, cb_s + bo, accum);
        run_attention(ln_g + (i + 1) * 256, ln_b + (i + 1) * 256);
    }

    // ---- pool + encoder + decode ----
    pool_gfeat<<<Bb, 256, 0, stream>>>(hbuf, gc, sl, gfeat);
    encoder_fc<<<Bb, 64, 0, stream>>>(gfeat, enc_w, enc_b, encb);
    lstm_decode<<<Bb, 256, 0, stream>>>(encb, emb, w_ih, w_hh, b_ih, b_hh, fc_w, fc_b, out);
}
